// SoftTree_42872363548690
// MI455X (gfx1250) — matrix operand order repaired and checked by reading, not compiled, not run on hardware
//
#include <hip/hip_runtime.h>
#include <math.h>

// ---------------- types ----------------
typedef __attribute__((ext_vector_type(16))) __bf16 v16bf;
typedef __attribute__((ext_vector_type(8)))  __bf16 v8bf;
typedef __attribute__((ext_vector_type(8)))  float  v8f;

#define D        1024
#define NLEAF    16
#define NINT     15
#define BM       128          // block rows
#define BN       64           // block cols
#define BK       32           // k per step (one bf16 WMMA depth)
#define LDK      (BK + 8)     // 80B row stride: 16B-aligned, bank-spread
#define OUTW     (NLEAF + D)  // 1040

__device__ __forceinline__ v16bf frag16(const __bf16* p0, const __bf16* p1) {
    v8bf lo = *(const v8bf*)p0;
    v8bf hi = *(const v8bf*)p1;
    return __builtin_shufflevector(lo, hi, 0,1,2,3,4,5,6,7,8,9,10,11,12,13,14,15);
}

// ---------------- bucketing kernels ----------------
__global__ void k_init(int* counts, int* cursors) {
    int t = threadIdx.x;
    if (t < NLEAF) { counts[t] = 0; cursors[t] = 0; }
}

__global__ void k_hist(const int* __restrict__ leaf_idx, int* counts, int N) {
    int i = blockIdx.x * blockDim.x + threadIdx.x;
    if (i < N) atomicAdd(&counts[leaf_idx[i]], 1);
}

__global__ void k_scan(const int* __restrict__ counts, int* offs) {
    if (threadIdx.x == 0) {
        int acc = 0;
        offs[0] = 0;
        for (int i = 0; i < NLEAF; ++i) {
            int p = (counts[i] + (BM - 1)) & ~(BM - 1);  // pad bucket to tile multiple
            acc += p;
            offs[i + 1] = acc;
        }
    }
}

__global__ void k_fill(int* perm, int cap) {
    int i = blockIdx.x * blockDim.x + threadIdx.x;
    if (i < cap) perm[i] = -1;
}

__global__ void k_scatter(const int* __restrict__ leaf_idx, const int* __restrict__ offs,
                          int* cursors, int* perm, int N) {
    int i = blockIdx.x * blockDim.x + threadIdx.x;
    if (i < N) {
        int l = leaf_idx[i];
        int pos = atomicAdd(&cursors[l], 1);
        perm[offs[l] + pos] = i;
    }
}

// ---------------- router: leaf log-probs (out[:, 0:16]) ----------------
__global__ __launch_bounds__(256)
void k_router(const float* __restrict__ xs, const float* __restrict__ weights,
              const float* __restrict__ biases, float* __restrict__ out, int N) {
    const int wave = threadIdx.x >> 5;
    const int lane = threadIdx.x & 31;
    const int row  = blockIdx.x * 8 + wave;
    if (row >= N) return;

    float acc[NINT];
#pragma unroll
    for (int j = 0; j < NINT; ++j) acc[j] = 0.f;

    const float* xrow = xs + (size_t)row * D;
    for (int k = lane; k < D; k += 32) {
        float x = xrow[k];
#pragma unroll
        for (int j = 0; j < NINT; ++j)
            acc[j] = fmaf(x, weights[j * D + k], acc[j]);
    }
#pragma unroll
    for (int j = 0; j < NINT; ++j) {
        for (int off = 16; off > 0; off >>= 1)
            acc[j] += __shfl_xor(acc[j], off, 32);
        acc[j] += biases[j];
    }
    if (lane < NLEAF) {
        int leaf = lane, node = 0;
        float s = 0.f;
        for (int d = 0; d < 4; ++d) {
            int bit = (leaf >> (3 - d)) & 1;
            float z = acc[node];
            if (!bit) z = -z;
            // stable logsigmoid(z) = min(z,0) - log1p(exp(-|z|))
            s += fminf(z, 0.f) - log1pf(__expf(-fabsf(z)));
            node = 2 * node + 1 + bit;
        }
        out[(size_t)row * OUTW + leaf] = s;
    }
}

// ---------------- routed expert GEMM (out[:, 16:1040]) ----------------
// 128x64 block tile, 8 wave32s, each wave owns a 32x32 sub-tile:
// 2 A-frags x 2 B-frags -> 4 v_wmma_f32_16x16x32_bf16 per K-step.
__global__ __launch_bounds__(256)
void k_leaf_gemm(const float* __restrict__ xs, const float* __restrict__ leafW,
                 const float* __restrict__ leafb, const int* __restrict__ perm,
                 const int* __restrict__ offs, float* __restrict__ out) {
    const int rowStart = blockIdx.x * BM;
    const int total = offs[NLEAF];
    if (rowStart >= total) return;

    // which leaf owns this tile (buckets are BM-padded, tiles never straddle)
    int leaf = NLEAF - 1;
#pragma unroll
    for (int i = 0; i < NLEAF; ++i)
        if (rowStart < offs[i + 1]) { leaf = i; break; }

    const int colStart = blockIdx.y * BN;
    const float* W = leafW + (size_t)leaf * D * D;

    __shared__ __align__(16) __bf16 As[BM][LDK];  // [row][k]
    __shared__ __align__(16) __bf16 Bs[BN][LDK];  // [col][k]  (N-major so K runs contiguous)

    const int tid  = threadIdx.x;
    const int lane = tid & 31;
    const int wave = tid >> 5;
    const int wr   = wave & 3;   // 32-row strip (4 strips x 32 = 128)
    const int wc   = wave >> 2;  // 32-col strip (2 strips x 32 = 64)
    const int hl   = lane >> 4;  // lane half -> K group
    const int l16  = lane & 15;

    v8f c00 = {}, c01 = {}, c10 = {}, c11 = {};

    // staging maps (fixed per thread)
    const int ar = tid >> 1;              // A: row 0..127
    const int ac = (tid & 1) * 16;        // A: 16 consecutive k
    const int bn = tid & 63;              // B: one column n
    const int bkq = (tid >> 6) * 8;       // B: 8 consecutive k

    for (int kk = 0; kk < D; kk += BK) {
        // ---- stage A tile: gathered rows, 16 k per thread, 2x b128 LDS stores ----
        {
            const int grow = perm[rowStart + ar];
            float4 f0 = make_float4(0.f,0.f,0.f,0.f), f1 = f0, f2 = f0, f3 = f0;
            if (grow >= 0) {
                const float* src = xs + (size_t)grow * D + kk + ac;
                f0 = *(const float4*)(src);
                f1 = *(const float4*)(src + 4);
                f2 = *(const float4*)(src + 8);
                f3 = *(const float4*)(src + 12);
                if (kk + BK < D) __builtin_prefetch(src + BK, 0, 0);
            }
            v8bf p0, p1;
            p0[0]=(__bf16)f0.x; p0[1]=(__bf16)f0.y; p0[2]=(__bf16)f0.z; p0[3]=(__bf16)f0.w;
            p0[4]=(__bf16)f1.x; p0[5]=(__bf16)f1.y; p0[6]=(__bf16)f1.z; p0[7]=(__bf16)f1.w;
            p1[0]=(__bf16)f2.x; p1[1]=(__bf16)f2.y; p1[2]=(__bf16)f2.z; p1[3]=(__bf16)f2.w;
            p1[4]=(__bf16)f3.x; p1[5]=(__bf16)f3.y; p1[6]=(__bf16)f3.z; p1[7]=(__bf16)f3.w;
            *(v8bf*)&As[ar][ac]     = p0;
            *(v8bf*)&As[ar][ac + 8] = p1;
        }
        // ---- stage B tile transposed: 8 k for one n, packed single b128 store ----
        {
            const float* wsrc = W + (size_t)(kk + bkq) * D + colStart + bn;
            v8bf p;
#pragma unroll
            for (int i = 0; i < 8; ++i)
                p[i] = (__bf16)wsrc[(size_t)i * D];
            *(v8bf*)&Bs[bn][bkq] = p;
        }
        __syncthreads();

        // ---- fragments per ISA 16-bit layouts ----
        // A 16x32: lanes 0-15 -> K {0..7,16..23}; lanes 16-31 -> K {8..15,24..31}
        const __bf16* a0r = &As[wr * 32 + l16][0];
        const __bf16* a1r = &As[wr * 32 + 16 + l16][0];
        v16bf a0 = frag16(a0r + hl * 8, a0r + hl * 8 + 16);
        v16bf a1 = frag16(a1r + hl * 8, a1r + hl * 8 + 16);
        // B 32x16: lane N = l16, lanes 0-15 -> K {0..7,16..23}; lanes 16-31 -> K {8..15,24..31} (N-major LDS)
        const __bf16* b0r = &Bs[wc * 32 + l16][0];
        const __bf16* b1r = &Bs[wc * 32 + 16 + l16][0];
        v16bf b0 = frag16(b0r + hl * 8, b0r + hl * 8 + 16);
        v16bf b1 = frag16(b1r + hl * 8, b1r + hl * 8 + 16);

        c00 = __builtin_amdgcn_wmma_f32_16x16x32_bf16(false, a0, false, b0, (short)0, c00, false, false);
        c01 = __builtin_amdgcn_wmma_f32_16x16x32_bf16(false, a0, false, b1, (short)0, c01, false, false);
        c10 = __builtin_amdgcn_wmma_f32_16x16x32_bf16(false, a1, false, b0, (short)0, c10, false, false);
        c11 = __builtin_amdgcn_wmma_f32_16x16x32_bf16(false, a1, false, b1, (short)0, c11, false, false);
        __syncthreads();
    }

    // ---- epilogue: scatter back + bias ----
    const int gcol0 = colStart + wc * 32 + l16;
    const int gcol1 = gcol0 + 16;
    const float bias0 = leafb[leaf * D + gcol0];
    const float bias1 = leafb[leaf * D + gcol1];
#pragma unroll
    for (int e = 0; e < 8; ++e) {
        const int m = hl * 8 + e;                        // C layout: VGPR e, lane half -> M
        const int tr0 = rowStart + wr * 32 + m;          // first 16-row strip
        const int tr1 = tr0 + 16;                        // second 16-row strip
        const int g0 = perm[tr0];
        const int g1 = perm[tr1];
        if (g0 >= 0) {
            float* orow = out + (size_t)g0 * OUTW + NLEAF;
            orow[gcol0] = c00[e] + bias0;
            orow[gcol1] = c01[e] + bias1;
        }
        if (g1 >= 0) {
            float* orow = out + (size_t)g1 * OUTW + NLEAF;
            orow[gcol0] = c10[e] + bias0;
            orow[gcol1] = c11[e] + bias1;
        }
    }
}

// ---------------- launch ----------------
extern "C" void kernel_launch(void* const* d_in, const int* in_sizes, int n_in,
                              void* d_out, int out_size, void* d_ws, size_t ws_size,
                              hipStream_t stream) {
    const float* xs       = (const float*)d_in[0];
    const int*   leaf_idx = (const int*)  d_in[1];
    const float* weights  = (const float*)d_in[2];
    const float* biases   = (const float*)d_in[3];
    // d_in[4] = masks: path structure recomputed analytically in k_router
    const float* leaf_W   = (const float*)d_in[5];
    const float* leaf_b   = (const float*)d_in[6];
    float*       out      = (float*)d_out;

    const int N = in_sizes[1];               // batch = #leaf indices

    int* wsi     = (int*)d_ws;
    int* offs    = wsi;                      // [17]
    int* counts  = wsi + 32;                 // [16]
    int* cursors = wsi + 64;                 // [16]
    int* perm    = wsi + 128;                // [N + NLEAF*BM]
    const int permCap = N + NLEAF * BM;

    k_init   <<<1, 64, 0, stream>>>(counts, cursors);
    k_hist   <<<(N + 255) / 256, 256, 0, stream>>>(leaf_idx, counts, N);
    k_scan   <<<1, 32, 0, stream>>>(counts, offs);
    k_fill   <<<(permCap + 255) / 256, 256, 0, stream>>>(perm, permCap);
    k_scatter<<<(N + 255) / 256, 256, 0, stream>>>(leaf_idx, offs, cursors, perm, N);

    k_router <<<(N + 7) / 8, 256, 0, stream>>>(xs, weights, biases, out, N);

    const int maxRowTiles = (N + BM - 1) / BM + NLEAF;  // worst case after padding
    dim3 grid(maxRowTiles, D / BN);
    k_leaf_gemm<<<grid, 256, 0, stream>>>(xs, leaf_W, leaf_b, perm, offs, out);
}
